// MultiheadAttention_80874234183732
// MI455X (gfx1250) — compile-verified, matrix operand order checked by reading, not run on hardware
//
#include <hip/hip_runtime.h>

// ---------------------------------------------------------------------------
// MI455X (gfx1250) multi-head attention forward.
// bf16 WMMA (v_wmma_f32_16x16x32_bf16) for all GEMMs, f32 accumulate.
// B=4, S=2048, D=768, H=12, DK=64.
// ---------------------------------------------------------------------------

typedef __bf16 bf16_t;
typedef __attribute__((ext_vector_type(16))) __bf16 v16bf;
typedef __attribute__((ext_vector_type(8)))  float  v8f;

#define BB 4
#define SS 2048
#define DD 768
#define HH 12
#define DKK 64
#define LSTR 2049   // padded LDS row stride (floats) -> conflict-free rows

union BFVec {
    v16bf v;
    bf16_t b[16];
    uint4 q[2];
};

__device__ inline v8f zero8() {
    v8f z;
#pragma unroll
    for (int i = 0; i < 8; ++i) z[i] = 0.0f;
    return z;
}

// Build a 16x32 bf16 A/B WMMA operand from a row-major row pointer.
// Per ISA layout: lane L holds row (L&15); K values = half*8+{0..7} and
// 16+half*8+{0..7}, half = L>>4.  Two 16B contiguous loads.
__device__ inline v16bf load_ab(const bf16_t* __restrict__ rowptr, int half) {
    BFVec r;
    r.q[0] = *(const uint4*)(rowptr + (half << 3));
    r.q[1] = *(const uint4*)(rowptr + 16 + (half << 3));
    return r.v;
}

__device__ inline v8f wmma_bf16(v16bf a, v16bf b, v8f c) {
    // (neg_a, A, neg_b, B, c_mod, C, reuse_a, reuse_b)
    return __builtin_amdgcn_wmma_f32_16x16x32_bf16(false, a, false, b,
                                                   (short)0, c, false, false);
}

// ---------------------------------------------------------------------------
// Stage 0: f32 -> bf16 conversion
// ---------------------------------------------------------------------------
__global__ __launch_bounds__(256) void mha_cvt_bf16(const float* __restrict__ src,
                                                    bf16_t* __restrict__ dst, int n) {
    int i = blockIdx.x * 256 + threadIdx.x;
    if (i < n) dst[i] = (bf16_t)src[i];
}

// ---------------------------------------------------------------------------
// Stage 1: QKV projection.  out = z @ W^T + bias.
// 32x32 register tile per wave (2x2 WMMA, 4 accumulators): each operand row
// is reused by two WMMAs.  blockIdx.y = 0/1/2 selects Q/K/V.
// Q,K stored [b,h,s,dk]; V stored transposed [b,h,dk,s] so the attention
// PV B-operand is a contiguous row-major load.
// ---------------------------------------------------------------------------
__global__ __launch_bounds__(256) void mha_qkv_gemm(
    const bf16_t* __restrict__ zbf,
    const bf16_t* __restrict__ wq, const bf16_t* __restrict__ wk,
    const bf16_t* __restrict__ wv,
    const float* __restrict__ bq, const float* __restrict__ bk,
    const float* __restrict__ bv,
    bf16_t* __restrict__ Qbf, bf16_t* __restrict__ Kbf,
    bf16_t* __restrict__ Vtbf) {
    const int proj = blockIdx.y;
    const bf16_t* W    = (proj == 0) ? wq : (proj == 1) ? wk : wv;
    const float*  bias = (proj == 0) ? bq : (proj == 1) ? bk : bv;

    const int wave = threadIdx.x >> 5;
    const int lane = threadIdx.x & 31;
    const int row  = lane & 15;
    const int half = lane >> 4;

    const int tile = blockIdx.x * 8 + wave;   // (8192/32)*(768/32) = 6144 tiles
    const int m0 = (tile / (DD / 32)) * 32;
    const int n0 = (tile % (DD / 32)) * 32;

    const bf16_t* arow0 = zbf + (long)(m0 + row) * DD;
    const bf16_t* arow1 = arow0 + 16 * DD;
    const bf16_t* brow0 = W + (long)(n0 + row) * DD;
    const bf16_t* brow1 = brow0 + 16 * DD;

    v8f acc00 = zero8(), acc01 = zero8(), acc10 = zero8(), acc11 = zero8();
#pragma unroll 2
    for (int k = 0; k < DD; k += 32) {
        __builtin_prefetch(arow0 + k + 64, 0, 0);
        __builtin_prefetch(brow0 + k + 64, 0, 0);
        const v16bf a0 = load_ab(arow0 + k, half);
        const v16bf a1 = load_ab(arow1 + k, half);
        const v16bf b0 = load_ab(brow0 + k, half);
        const v16bf b1 = load_ab(brow1 + k, half);
        acc00 = wmma_bf16(a0, b0, acc00);
        acc01 = wmma_bf16(a0, b1, acc01);
        acc10 = wmma_bf16(a1, b0, acc10);
        acc11 = wmma_bf16(a1, b1, acc11);
    }

#pragma unroll
    for (int nt = 0; nt < 2; ++nt) {
        const int n = n0 + nt * 16 + row;   // output column (D-dim)
        const float bsv = bias[n];
        const int h = n >> 6;               // head
        const int d = n & 63;               // dk index
#pragma unroll
        for (int mt = 0; mt < 2; ++mt) {
            const v8f acc = (mt == 0) ? (nt == 0 ? acc00 : acc01)
                                      : (nt == 0 ? acc10 : acc11);
#pragma unroll
            for (int r = 0; r < 8; ++r) {
                const int m = m0 + mt * 16 + r + 8 * half;  // row (b*S + s)
                const int bidx = m >> 11;
                const int s    = m & (SS - 1);
                const float val = acc[r] + bsv;
                const long bh = (long)bidx * HH + h;
                if (proj == 0)      Qbf[(bh * SS + s) * DKK + d] = (bf16_t)val;
                else if (proj == 1) Kbf[(bh * SS + s) * DKK + d] = (bf16_t)val;
                else                Vtbf[(bh * DKK + d) * SS + s] = (bf16_t)val;
            }
        }
    }
}

// ---------------------------------------------------------------------------
// Stage 2: attention.  One block per (b, 16-query tile); 8 waves.
// LDS: logits[16][2049] + avg[16][2049] + Opart[8][16][64] + reductions
// (~289 KB of the WGP's 320 KB).  Loops all 12 heads so the cross-head avg
// accumulates in LDS and the 67 MB avg_weights output is written once;
// the 805 MB [b,h,s,s] logit tensor never touches HBM.
// The softmax-sum pass overwrites logits with exp(l-m), so the PV pass only
// converts to bf16 (no second exp sweep).
// ---------------------------------------------------------------------------
__global__ __launch_bounds__(256) void mha_attn(
    const bf16_t* __restrict__ Qbf, const bf16_t* __restrict__ Kbf,
    const bf16_t* __restrict__ Vtbf,
    bf16_t* __restrict__ ctxbf, float* __restrict__ avg_out) {
    extern __shared__ __align__(16) char smem_raw[];
    float* logits = (float*)smem_raw;            // 16*LSTR
    float* avg    = logits + 16 * LSTR;          // 16*LSTR
    float* Opart  = avg + 16 * LSTR;             // 8*16*64
    float* red    = Opart + 8 * 16 * 64;         // 16*16
    float* rowmax = red + 256;                   // 16
    float* rowrcp = rowmax + 16;                 // 16

    const int b  = blockIdx.y;
    const int q0 = blockIdx.x * 16;
    const int tid  = threadIdx.x;
    const int wave = tid >> 5;
    const int lane = tid & 31;
    const int row  = lane & 15;
    const int half = lane >> 4;

    for (int h = 0; h < HH; ++h) {
        const long bh = (long)b * HH + h;
        const bf16_t* Qbase  = Qbf  + (bh * SS + q0) * DKK;
        const bf16_t* Kbase  = Kbf  + bh * SS * DKK;
        const bf16_t* Vtbase = Vtbf + bh * DKK * SS;

        // Q tile A-operands (16 rows x 64, two K=32 chunks), held in VGPRs.
        const v16bf aq0 = load_ab(Qbase + row * DKK, half);
        const v16bf aq1 = load_ab(Qbase + row * DKK + 32, half);

        // ---- QK^T: 128 key tiles split across 8 waves -------------------
        for (int jt = wave; jt < SS / 16; jt += 8) {
            const int kb = jt * 16;
            const bf16_t* Krow = Kbase + (kb + row) * DKK;  // B: col n = key
            v8f acc = zero8();
            acc = wmma_bf16(aq0, load_ab(Krow, half), acc);
            acc = wmma_bf16(aq1, load_ab(Krow + 32, half), acc);
#pragma unroll
            for (int r = 0; r < 8; ++r) {
                const int qm = r + 8 * half;
                const float l = acc[r] * 0.125f;        // 1/sqrt(DK)
                const int idx = qm * LSTR + kb + row;
                logits[idx] = l;
                avg[idx] = (h == 0 ? 0.0f : avg[idx]) + l;
            }
        }
        __syncthreads();

        // ---- softmax row stats (16 threads per row) ---------------------
        {
            const int rr = tid >> 4;
            const int cc = tid & 15;
            float m = -1e30f;
            for (int k = cc; k < SS; k += 16)
                m = fmaxf(m, logits[rr * LSTR + k]);
            red[rr * 16 + cc] = m;
            __syncthreads();
            if (cc == 0) {
                float mm = red[rr * 16];
#pragma unroll
                for (int i = 1; i < 16; ++i) mm = fmaxf(mm, red[rr * 16 + i]);
                rowmax[rr] = mm;
            }
            __syncthreads();
            const float mm = rowmax[rr];
            float s = 0.0f;
            for (int k = cc; k < SS; k += 16) {
                const float e = __expf(logits[rr * LSTR + k] - mm);
                logits[rr * LSTR + k] = e;      // raw logit dead: cache prob
                s += e;
            }
            red[rr * 16 + cc] = s;
            __syncthreads();
            if (cc == 0) {
                float ss = 0.0f;
#pragma unroll
                for (int i = 0; i < 16; ++i) ss += red[rr * 16 + i];
                rowrcp[rr] = 1.0f / ss;
            }
            __syncthreads();
        }

        // ---- P @ V: each wave owns 256 keys -----------------------------
        v8f o0 = zero8(), o1 = zero8(), o2 = zero8(), o3 = zero8();
        const int kstart = wave * (SS / 8);
        for (int kb = kstart; kb < kstart + SS / 8; kb += 32) {
            BFVec ap;
            const float* prow = logits + row * LSTR + kb + half * 8;
#pragma unroll
            for (int j = 0; j < 8; ++j) {
                ap.b[j]     = (bf16_t)prow[j];
                ap.b[j + 8] = (bf16_t)prow[16 + j];
            }
            const v16bf a = ap.v;
            o0 = wmma_bf16(a, load_ab(Vtbase + (row)      * SS + kb, half), o0);
            o1 = wmma_bf16(a, load_ab(Vtbase + (16 + row) * SS + kb, half), o1);
            o2 = wmma_bf16(a, load_ab(Vtbase + (32 + row) * SS + kb, half), o2);
            o3 = wmma_bf16(a, load_ab(Vtbase + (48 + row) * SS + kb, half), o3);
        }
#pragma unroll
        for (int r = 0; r < 8; ++r) {
            const int qm = r + 8 * half;
            float* op = Opart + (wave * 16 + qm) * 64 + row;
            op[0]  = o0[r];
            op[16] = o1[r];
            op[32] = o2[r];
            op[48] = o3[r];
        }
        __syncthreads();

        // ---- cross-wave reduce, normalize, store ctx (bf16) -------------
        for (int e = tid; e < 16 * 64; e += 256) {
            const int qm = e >> 6;
            const int d  = e & 63;
            float s = 0.0f;
#pragma unroll
            for (int w = 0; w < 8; ++w) s += Opart[(w * 16 + qm) * 64 + d];
            s *= rowrcp[qm];
            ctxbf[(long)(b * SS + q0 + qm) * DD + h * DKK + d] = (bf16_t)s;
        }
        __syncthreads();
    }

    // ---- write averaged pre-softmax weights (once per block) ------------
    const float inv_h = 1.0f / (float)HH;
    for (int e = tid; e < 16 * SS; e += 256) {
        const int qm = e >> 11;
        const int k  = e & (SS - 1);
        avg_out[(long)(b * SS + q0 + qm) * SS + k] = avg[qm * LSTR + k] * inv_h;
    }
}

// ---------------------------------------------------------------------------
// Stage 3: final FC.  out = ctx @ fc_w^T + fc_b  (f32 output)
// 32x32 register tile per wave (2x2 WMMA).
// ---------------------------------------------------------------------------
__global__ __launch_bounds__(256) void mha_fc_gemm(
    const bf16_t* __restrict__ ctx, const bf16_t* __restrict__ W,
    const float* __restrict__ bias, float* __restrict__ out) {
    const int wave = threadIdx.x >> 5;
    const int lane = threadIdx.x & 31;
    const int row  = lane & 15;
    const int half = lane >> 4;

    const int tile = blockIdx.x * 8 + wave;
    const int m0 = (tile / (DD / 32)) * 32;
    const int n0 = (tile % (DD / 32)) * 32;

    const bf16_t* arow0 = ctx + (long)(m0 + row) * DD;
    const bf16_t* arow1 = arow0 + 16 * DD;
    const bf16_t* brow0 = W + (long)(n0 + row) * DD;
    const bf16_t* brow1 = brow0 + 16 * DD;

    v8f acc00 = zero8(), acc01 = zero8(), acc10 = zero8(), acc11 = zero8();
#pragma unroll 2
    for (int k = 0; k < DD; k += 32) {
        __builtin_prefetch(arow0 + k + 64, 0, 0);
        __builtin_prefetch(brow0 + k + 64, 0, 0);
        const v16bf a0 = load_ab(arow0 + k, half);
        const v16bf a1 = load_ab(arow1 + k, half);
        const v16bf b0 = load_ab(brow0 + k, half);
        const v16bf b1 = load_ab(brow1 + k, half);
        acc00 = wmma_bf16(a0, b0, acc00);
        acc01 = wmma_bf16(a0, b1, acc01);
        acc10 = wmma_bf16(a1, b0, acc10);
        acc11 = wmma_bf16(a1, b1, acc11);
    }

#pragma unroll
    for (int nt = 0; nt < 2; ++nt) {
        const int n = n0 + nt * 16 + row;
        const float bsv = bias[n];
#pragma unroll
        for (int mt = 0; mt < 2; ++mt) {
            const v8f acc = (mt == 0) ? (nt == 0 ? acc00 : acc01)
                                      : (nt == 0 ? acc10 : acc11);
#pragma unroll
            for (int r = 0; r < 8; ++r) {
                const int m = m0 + mt * 16 + r + 8 * half;
                out[(long)m * DD + n] = acc[r] + bsv;
            }
        }
    }
}

// ---------------------------------------------------------------------------
extern "C" void kernel_launch(void* const* d_in, const int* in_sizes, int n_in,
                              void* d_out, int out_size, void* d_ws, size_t ws_size,
                              hipStream_t stream) {
    (void)in_sizes; (void)n_in; (void)out_size; (void)ws_size;

    const float* z    = (const float*)d_in[0];
    const float* wq_w = (const float*)d_in[1];
    const float* wq_b = (const float*)d_in[2];
    const float* wk_w = (const float*)d_in[3];
    const float* wk_b = (const float*)d_in[4];
    const float* wv_w = (const float*)d_in[5];
    const float* wv_b = (const float*)d_in[6];
    const float* fc_w = (const float*)d_in[7];
    const float* fc_b = (const float*)d_in[8];

    float* out_attn = (float*)d_out;                       // [B,S,D]
    float* out_avg  = out_attn + (size_t)BB * SS * DD;     // [B,S,S]

    const size_t nzd = (size_t)BB * SS * DD;   // 6,291,456
    const size_t nw  = (size_t)DD * DD;        //   589,824

    char* ws = (char*)d_ws;
    bf16_t* zbf  = (bf16_t*)ws;  ws += nzd * 2;
    bf16_t* wqbf = (bf16_t*)ws;  ws += nw * 2;
    bf16_t* wkbf = (bf16_t*)ws;  ws += nw * 2;
    bf16_t* wvbf = (bf16_t*)ws;  ws += nw * 2;
    bf16_t* fcbf = (bf16_t*)ws;  ws += nw * 2;
    bf16_t* Qbf  = (bf16_t*)ws;  ws += nzd * 2;
    bf16_t* Kbf  = (bf16_t*)ws;  ws += nzd * 2;
    bf16_t* Vtbf = (bf16_t*)ws;  ws += nzd * 2;
    bf16_t* ctxbf= (bf16_t*)ws;  ws += nzd * 2;

    // Stage 0: conversions
    mha_cvt_bf16<<<(int)((nzd + 255) / 256), 256, 0, stream>>>(z, zbf, (int)nzd);
    mha_cvt_bf16<<<(int)((nw + 255) / 256), 256, 0, stream>>>(wq_w, wqbf, (int)nw);
    mha_cvt_bf16<<<(int)((nw + 255) / 256), 256, 0, stream>>>(wk_w, wkbf, (int)nw);
    mha_cvt_bf16<<<(int)((nw + 255) / 256), 256, 0, stream>>>(wv_w, wvbf, (int)nw);
    mha_cvt_bf16<<<(int)((nw + 255) / 256), 256, 0, stream>>>(fc_w, fcbf, (int)nw);

    // Stage 1: QKV projection (6144 wave-tiles / 8 waves = 768 blocks, x3)
    mha_qkv_gemm<<<dim3(768, 3), 256, 0, stream>>>(
        zbf, wqbf, wkbf, wvbf, wq_b, wk_b, wv_b, Qbf, Kbf, Vtbf);

    // Stage 2: attention.  Dynamic LDS ~289 KB (CDNA5 WGP has 320 KB).
    const size_t smem = (size_t)(16 * LSTR * 2 + 8 * 16 * 64 + 256 + 32) * sizeof(float);
    mha_attn<<<dim3(SS / 16, BB), 256, smem, stream>>>(Qbf, Kbf, Vtbf, ctxbf, out_avg);

    // Stage 3: final FC
    mha_fc_gemm<<<768, 256, 0, stream>>>(ctxbf, fcbf, fc_b, out_attn);
}
